// DRA_52905407152670
// MI455X (gfx1250) — matrix operand order repaired and checked by reading, not compiled, not run on hardware
//
#include <hip/hip_runtime.h>
#include <hip/hip_bf16.h>
#include <math.h>

// ---------------------------------------------------------------------------
// Types for CDNA5 WMMA
// ---------------------------------------------------------------------------
typedef __bf16 bf16_t;
typedef __attribute__((ext_vector_type(16))) __bf16 v16bf;
typedef __attribute__((ext_vector_type(8)))  __bf16 v8bf;
typedef __attribute__((ext_vector_type(4)))  __bf16 v4bf;
typedef __attribute__((ext_vector_type(8)))  float  v8f;

static __device__ __forceinline__ bf16_t f2bf(float f) { return (bf16_t)f; }

// LDS row stride (bf16 elems): 48 = 96 bytes (16B multiple -> aligned b128
// loads; 24-bank stagger avoids heavy conflicts on fragment fetch).
#define LSTR 48

// ---------------------------------------------------------------------------
// Generic bf16-WMMA GEMM:  C[M,N] = A[M,K] * B
//   B element (k,n) = transB ? Bm[n*ldb + k] : Bm[k*ldb + n]
// Tile: 64 (M) x 128 (N) per workgroup, 256 threads = 8 waves.
// Wave (wm = wave&3, wn = wave>>2) computes rows [wm*16, +16) x cols
// [wn*64, +64) via 4 f32 accumulators; K stepped by 32 through LDS.
// A staged row-major [m][k]; B staged TRANSPOSED [n][k] so both A and B
// fragments are 2x contiguous 16B chunks per lane -> ds_load_b128.
// Dims used here: M in {64, 32768}, N in {768, 2304}, K = 768 — all tile-
// aligned, no bounds checks.
// ---------------------------------------------------------------------------
__global__ __launch_bounds__(256)
void DRA_wmma_gemm(const float* __restrict__ A, const float* __restrict__ Bm,
                   float* __restrict__ C, int M, int N, int K,
                   int lda, int ldb, int ldc, int transB)
{
    __shared__ __align__(16) bf16_t lA[64][LSTR];    //  6 KB
    __shared__ __align__(16) bf16_t lBt[128][LSTR];  // 12 KB (B transposed)

    const int tid  = threadIdx.x;
    const int lane = tid & 31;
    const int wave = tid >> 5;
    const int wm   = wave & 3;          // row strip 0..3 (16 rows each)
    const int wn   = wave >> 2;         // col half 0..1 (64 cols each)
    const int tileM = blockIdx.y * 64;
    const int tileN = blockIdx.x * 128;

    v8f acc[4] = {v8f{}, v8f{}, v8f{}, v8f{}};

    const int mfr = lane & 15;          // fragment row within strip
    const int kh  = (lane >> 4) << 3;   // 0 or 8 (A frag k base)
    const int kg  = (lane >> 4) << 4;   // 0 or 16 (unused; B frags also use kh)

    for (int k0 = 0; k0 < K; k0 += 32) {
        // ---- prefetch next K tile (speculative, RT) ----
        if (k0 + 32 < K) {
            const int pr = tid & 63;
            __builtin_prefetch(A + (size_t)(tileM + pr) * lda + (k0 + 32), 0, 0);
            if (transB)
                __builtin_prefetch(Bm + (size_t)(tileN + (tid & 127)) * ldb + (k0 + 32), 0, 0);
            else
                __builtin_prefetch(Bm + (size_t)(k0 + 32 + (tid & 31)) * ldb + tileN, 0, 0);
        }

        // ---- stage A tile 64x32 : float4 coalesced reads, v4bf LDS stores
        for (int i = tid; i < 512; i += 256) {          // 512 float4 chunks
            int r = i >> 3, c4 = (i & 7) << 2;
            const float4 v = *reinterpret_cast<const float4*>(
                A + (size_t)(tileM + r) * lda + (k0 + c4));
            v4bf p = { f2bf(v.x), f2bf(v.y), f2bf(v.z), f2bf(v.w) };
            *reinterpret_cast<v4bf*>(&lA[r][c4]) = p;
        }
        // ---- stage B tile 32x128 transposed into lBt[n][k]
        if (transB) {
            // B(k,n) = Bm[n*ldb + k]: coalesced float4 over k, vector LDS store
            for (int i = tid; i < 1024; i += 256) {
                int n = i >> 3, k4 = (i & 7) << 2;
                const float4 v = *reinterpret_cast<const float4*>(
                    Bm + (size_t)(tileN + n) * ldb + (k0 + k4));
                v4bf p = { f2bf(v.x), f2bf(v.y), f2bf(v.z), f2bf(v.w) };
                *reinterpret_cast<v4bf*>(&lBt[n][k4]) = p;
            }
        } else {
            // B(k,n) = Bm[k*ldb + n]: coalesced float4 over n, scalar transposed stores
            for (int i = tid; i < 1024; i += 256) {
                int k = i >> 5, n4 = (i & 31) << 2;
                const float4 v = *reinterpret_cast<const float4*>(
                    Bm + (size_t)(k0 + k) * ldb + (tileN + n4));
                lBt[n4 + 0][k] = f2bf(v.x);
                lBt[n4 + 1][k] = f2bf(v.y);
                lBt[n4 + 2][k] = f2bf(v.z);
                lBt[n4 + 3][k] = f2bf(v.w);
            }
        }
        __syncthreads();

        // ---- A fragment: two aligned 16B chunks -> 2x ds_load_b128
        union { v16bf v; v8bf h[2]; } af;
        {
            const int m = (wm << 4) + mfr;
            af.h[0] = *reinterpret_cast<const v8bf*>(&lA[m][kh]);
            af.h[1] = *reinterpret_cast<const v8bf*>(&lA[m][16 + kh]);
        }

        // ---- 4 B fragments + WMMA
        #pragma unroll
        for (int c = 0; c < 4; ++c) {
            union { v16bf v; v8bf h[2]; } bf;
            const int n = (wn << 6) + (c << 4) + mfr;
            bf.h[0] = *reinterpret_cast<const v8bf*>(&lBt[n][kh]);
            bf.h[1] = *reinterpret_cast<const v8bf*>(&lBt[n][16 + kh]);
            acc[c] = __builtin_amdgcn_wmma_f32_16x16x32_bf16(
                        false, af.v, false, bf.v, (short)0, acc[c], false, false);
        }
        __syncthreads();
    }

    // ---- store: C/D layout — VGPR r: M=r (lanes 0-15), M=8+r (lanes 16-31)
    const int mbase = tileM + (wm << 4) + ((lane >> 4) << 3);
    const int nbase = tileN + (wn << 6) + mfr;
    #pragma unroll
    for (int c = 0; c < 4; ++c) {
        #pragma unroll
        for (int r = 0; r < 8; ++r) {
            C[(size_t)(mbase + r) * ldc + nbase + (c << 4)] = acc[c][r];
        }
    }
}

// ---------------------------------------------------------------------------
// Fused attention scoring + softmax + masked-mean reweighted pooling.
// One workgroup (256 threads, 8 waves) per batch row b.
//   scores[s] = dot(tanh(Se[b,s,:] + hWd[b,:] + A[b,:]), w)
//   m = softmax(scores)            (unmasked softmax, as in reference)
//   at[b,:] = sum_s m[s]*mask[b,s]*X[b,s,:] / sum_s mask[b,s]
// ---------------------------------------------------------------------------
__global__ __launch_bounds__(256)
void DRA_attn_pool(const float* __restrict__ Se, const float* __restrict__ hWd,
                   const float* __restrict__ Av, const float* __restrict__ w,
                   const float* __restrict__ X, const float* __restrict__ mask,
                   float* __restrict__ at, int S, int H)
{
    const int b    = blockIdx.x;
    const int tid  = threadIdx.x;
    const int lane = tid & 31;
    const int wave = tid >> 5;

    __shared__ float sc[512];       // scores -> probs -> coeffs
    __shared__ float redm[8], reds[8], redd[8];

    const float* SeB = Se + (size_t)b * S * H;
    const float* XB  = X  + (size_t)b * S * H;
    const float* hb  = hWd + (size_t)b * H;
    const float* ab  = Av  + (size_t)b * H;
    const float* mb  = mask + (size_t)b * S;

    // ---- phase 1: scores (one wave per s, lanes stride H, wave32 reduce) ----
    for (int s = wave; s < S; s += 8) {
        const float* row = SeB + (size_t)s * H;
        float acc = 0.f;
        for (int h = lane; h < H; h += 32) {
            float mval = row[h] + hb[h] + ab[h];
            acc += tanhf(mval) * w[h];
        }
        #pragma unroll
        for (int off = 16; off > 0; off >>= 1)
            acc += __shfl_xor(acc, off, 32);
        if (lane == 0) sc[s] = acc;
    }
    __syncthreads();

    // ---- phase 2: softmax over S + mask denom ----
    float lmax = -3.4e38f;
    for (int s = tid; s < S; s += 256) lmax = fmaxf(lmax, sc[s]);
    #pragma unroll
    for (int off = 16; off > 0; off >>= 1)
        lmax = fmaxf(lmax, __shfl_xor(lmax, off, 32));
    if (lane == 0) redm[wave] = lmax;
    __syncthreads();
    float bmax = redm[0];
    #pragma unroll
    for (int i = 1; i < 8; ++i) bmax = fmaxf(bmax, redm[i]);
    __syncthreads();

    float lsum = 0.f, lden = 0.f;
    for (int s = tid; s < S; s += 256) {
        float e = __expf(sc[s] - bmax);
        sc[s] = e;
        lsum += e;
        lden += mb[s];
    }
    #pragma unroll
    for (int off = 16; off > 0; off >>= 1) {
        lsum += __shfl_xor(lsum, off, 32);
        lden += __shfl_xor(lden, off, 32);
    }
    if (lane == 0) { reds[wave] = lsum; redd[wave] = lden; }
    __syncthreads();
    float esum = 0.f, msum = 0.f;
    #pragma unroll
    for (int i = 0; i < 8; ++i) { esum += reds[i]; msum += redd[i]; }
    const float scale = 1.f / (esum * msum);
    __syncthreads();

    // coeff[s] = softmax * mask / denom
    for (int s = tid; s < S; s += 256) sc[s] = sc[s] * mb[s] * scale;
    __syncthreads();

    // ---- phase 3: pooling, coalesced over h ----
    for (int h = tid; h < H; h += 256) {
        float acc = 0.f;
        for (int s = 0; s < S; ++s)
            acc += sc[s] * XB[(size_t)s * H + h];
        at[(size_t)b * H + h] = acc;
    }
}

// ---------------------------------------------------------------------------
// GRU cell elementwise (torch gate order r,z,n):
//   r = sig(ir+hr); z = sig(iz+hz); n = tanh(in + r*hn); h' = (1-z)*n + z*h
// ---------------------------------------------------------------------------
__global__ __launch_bounds__(256)
void DRA_gru_cell(const float* __restrict__ gi, const float* __restrict__ gh,
                  const float* __restrict__ h, float* __restrict__ hout,
                  int B, int G)
{
    int idx = blockIdx.x * blockDim.x + threadIdx.x;
    if (idx >= B * G) return;
    int b = idx / G, g = idx - b * G;
    const float* gib = gi + (size_t)b * 3 * G;
    const float* ghb = gh + (size_t)b * 3 * G;
    float ir = gib[g], iz = gib[G + g], in_ = gib[2 * G + g];
    float hr = ghb[g], hz = ghb[G + g], hn  = ghb[2 * G + g];
    float r = 1.f / (1.f + __expf(-(ir + hr)));
    float z = 1.f / (1.f + __expf(-(iz + hz)));
    float n = tanhf(in_ + r * hn);
    hout[idx] = (1.f - z) * n + z * h[idx];
}

// ---------------------------------------------------------------------------
// Host orchestration
// ---------------------------------------------------------------------------
extern "C" void kernel_launch(void* const* d_in, const int* in_sizes, int n_in,
                              void* d_out, int out_size, void* d_ws, size_t ws_size,
                              hipStream_t stream) {
    const float* sent_rep  = (const float*)d_in[0];   // (B,H)
    const float* attn_mask = (const float*)d_in[1];   // (B,S)
    const float* aspect    = (const float*)d_in[2];   // (B,H)
    const float* X         = (const float*)d_in[3];   // (B,S,H)
    const float* ws        = (const float*)d_in[4];   // (H,H)
    const float* wa        = (const float*)d_in[5];   // (H,H)
    const float* w         = (const float*)d_in[6];   // (H)
    const float* whs       = (const float*)d_in[7];   // (H,G)
    const float* wd1       = (const float*)d_in[8];   // (H,H)
    const float* wd        = (const float*)d_in[9];   // (G,H)
    const float* w_ih      = (const float*)d_in[10];  // (3G,H)
    const float* w_hh      = (const float*)d_in[11];  // (3G,G)

    const int B = 64, S = 512, H = 768, G = 768;

    // scratch layout (floats)
    float* f   = (float*)d_ws;
    size_t o   = 0;
    float* Se  = f + o; o += (size_t)B * S * H;   // (B,S,H)  ~100.7 MB
    float* Av  = f + o; o += (size_t)B * H;
    float* hWd = f + o; o += (size_t)B * H;
    float* at  = f + o; o += (size_t)B * H;
    float* hA  = f + o; o += (size_t)B * G;
    float* hB  = f + o; o += (size_t)B * G;
    float* gi  = f + o; o += (size_t)B * 3 * G;
    float* gh  = f + o; o += (size_t)B * 3 * G;
    (void)o; (void)ws_size; (void)in_sizes; (void)n_in; (void)out_size;

    dim3 blk(256);

    // Se = X @ ws  (the 38.7 GFLOP GEMM)
    DRA_wmma_gemm<<<dim3(H / 128, (B * S) / 64), blk, 0, stream>>>(
        X, ws, Se, B * S, H, H, H, H, H, 0);
    // Av = aspect @ wa
    DRA_wmma_gemm<<<dim3(H / 128, 1), blk, 0, stream>>>(
        aspect, wa, Av, B, H, H, H, H, H, 0);
    // h = sentence_representation
    hipMemcpyAsync(hA, sent_rep, (size_t)B * H * sizeof(float),
                   hipMemcpyDeviceToDevice, stream);

    float* hcur = hA;
    float* hnxt = hB;
    for (int t = 0; t < 3; ++t) {
        const float* Wd = (t == 0) ? wd1 : wd;
        // hWd = h @ Wd
        DRA_wmma_gemm<<<dim3(H / 128, 1), blk, 0, stream>>>(
            hcur, Wd, hWd, B, H, H, H, H, H, 0);
        // at = pool(softmax(tanh(Se + hWd + Av) . w) * mask, X) / denom
        DRA_attn_pool<<<B, blk, 0, stream>>>(Se, hWd, Av, w, X, attn_mask, at, S, H);
        if (t == 0) {
            // h = h @ whs
            DRA_wmma_gemm<<<dim3(G / 128, 1), blk, 0, stream>>>(
                hcur, whs, hnxt, B, G, H, H, G, G, 0);
            float* tmp = hcur; hcur = hnxt; hnxt = tmp;
        }
        // gi = at @ w_ih.T   (B x 3G), w_ih is (3G,H) row-major -> transB, ldb=H
        DRA_wmma_gemm<<<dim3((3 * G) / 128, 1), blk, 0, stream>>>(
            at, w_ih, gi, B, 3 * G, H, H, H, 3 * G, 1);
        // gh = h @ w_hh.T    (B x 3G), w_hh is (3G,G) -> transB, ldb=G
        DRA_wmma_gemm<<<dim3((3 * G) / 128, 1), blk, 0, stream>>>(
            hcur, w_hh, gh, B, 3 * G, G, G, G, 3 * G, 1);
        // h' = GRU(at, h)
        DRA_gru_cell<<<(B * G + 255) / 256, blk, 0, stream>>>(gi, gh, hcur, hnxt, B, G);
        float* tmp = hcur; hcur = hnxt; hnxt = tmp;
    }

    hipMemcpyAsync(d_out, hcur, (size_t)B * G * sizeof(float),
                   hipMemcpyDeviceToDevice, stream);
}
